// XEyTransformerLayer_32246614458632
// MI455X (gfx1250) — matrix operand order repaired and checked by reading, not compiled, not run on hardware
//
#include <hip/hip_runtime.h>
#include <cstdint>
#include <cstddef>

// ---------------------------------------------------------------------------
// CDNA5 (gfx1250) wave32 WMMA implementation of the XEyTransformerLayer.
// Big GEMMs use v_wmma_f32_16x16x32_bf16 (f32 accumulate, bf16 inputs).
// The dominant E-streaming kernel stages its tiles into LDS with the
// Tensor Data Mover (tensor_load_to_lds + s_wait_tensorcnt).
// ---------------------------------------------------------------------------

typedef __attribute__((ext_vector_type(16))) __bf16 v16bf;
typedef __attribute__((ext_vector_type(8)))  float  v8f;
typedef __attribute__((ext_vector_type(4)))  unsigned uint4v;
typedef __attribute__((ext_vector_type(8)))  int      int8v;
typedef __attribute__((ext_vector_type(4)))  int      int4v;

#if defined(__has_builtin)
#if __has_builtin(__builtin_amdgcn_tensor_load_to_lds)
#define HAVE_TDM 1
#endif
#endif
#ifndef HAVE_TDM
#define HAVE_TDM 0
#endif

#define DEVI __device__ __forceinline__

DEVI v8f wmma_bf16(v16bf a, v16bf b, v8f c) {
  return __builtin_amdgcn_wmma_f32_16x16x32_bf16(false, a, false, b, (short)0, c,
                                                 false, false);
}

// A fragment (16x32, M on lanes): element e of lane L (m = L%16, hf = L/16)
//   k = 16*hf + e
DEVI v16bf ldA(const float* src, int m, int ld, int kk, int hf) {
  v16bf a;
  const float* row = src + (size_t)m * ld + kk + (hf << 4);
#pragma unroll
  for (int g2 = 0; g2 < 2; ++g2)
#pragma unroll
    for (int e = 0; e < 8; ++e)
      a[g2 * 8 + e] = (__bf16)row[g2 * 8 + e];
  return a;
}

DEVI v16bf ldA_h(const __bf16* src, int m, int ld, int kk, int hf) {
  v16bf a;
  const __bf16* row = src + (size_t)m * ld + kk + (hf << 4);
#pragma unroll
  for (int g2 = 0; g2 < 2; ++g2)
#pragma unroll
    for (int e = 0; e < 8; ++e)
      a[g2 * 8 + e] = row[g2 * 8 + e];
  return a;
}

// B fragment (32x16, N on lanes): element e of lane L (n = L%16, hf = L/16)
//   k = 16*hf + e   (contiguous chunk per lane-half, same k order as ldA)
// Source row-major over k: src[k*ld + n]
DEVI v16bf ldB(const float* src, int ncol, int ld, int kk, int hf) {
  v16bf b;
  const float* base = src + (size_t)(kk + (hf << 4)) * ld + ncol;
#pragma unroll
  for (int e = 0; e < 16; ++e) b[e] = (__bf16)base[(size_t)e * ld];
  return b;
}

// B fragment where the source is stored row-per-n (i.e. src[n*ld + k]):
DEVI v16bf ldBt(const float* src, int ncol, int ld, int kk, int hf) {
  v16bf b;
  const float* row = src + (size_t)ncol * ld + kk + (hf << 4);
#pragma unroll
  for (int e = 0; e < 16; ++e) b[e] = (__bf16)row[e];
  return b;
}

// ---------------------------------------------------------------------------
// yvec scratch layout (floats): ye_add[64] | ye_mul[64] | yx_add[2048] |
//                               yx_mul[2048] | tcos[8]
#define YV_YEA 0
#define YV_YEM 64
#define YV_YXA 128
#define YV_YXM 2176
#define YV_TC  4224

// ---------------------------------------------------------------------------
// Kernel: whole y path + y-derived modulation vectors.  grid=8, block=256.
__global__ __launch_bounds__(256) void k_ypath(
    const float* X, const float* E, const float* yIn, const float* t_init,
    const float* alphas,
    const float* Wyy, const float* byy, const float* Wxy, const float* bxy,
    const float* Wey, const float* bey, const float* Wo1, const float* bo1,
    const float* Wo2, const float* bo2, const float* g1, const float* be1,
    const float* Wl1, const float* bl1, const float* Wl2, const float* bl2,
    const float* g2, const float* be2,
    const float* Wyea, const float* byea, const float* Wyem, const float* byem,
    const float* Wyxa, const float* byxa, const float* Wyxm, const float* byxm,
    float* yo, float* yvec) {
  int b = blockIdx.x, t = threadIdx.x;
  __shared__ float fx[1024];
  __shared__ float fe[256];
  __shared__ float ny[64], tmp64[64], yrS[64], hS[2048];
  __shared__ float eS[256], eQ[256], eMn[256], eMx[256];
  __shared__ float mv2[2];

  const float* Xb = X + (size_t)b * 65536;
  {  // xtoy features: mean/min/max/std(ddof=1) over the 256 nodes, per channel
    float s = 0.f, q = 0.f, mn = 3.402823466e38f, mx = -3.402823466e38f;
    for (int i = 0; i < 256; ++i) {
      float v = Xb[(size_t)i * 256 + t];
      s += v; q += v * v; mn = fminf(mn, v); mx = fmaxf(mx, v);
    }
    float mu_ = s / 256.f;
    float var = (q - 256.f * mu_ * mu_) / 255.f;
    fx[t] = mu_; fx[256 + t] = mn; fx[512 + t] = mx;
    fx[768 + t] = sqrtf(fmaxf(var, 0.f));
  }
  {  // etoy features over 65536 positions, per channel (4-way split)
    const float* Eb = E + (size_t)b * 4194304;
    int c = t & 63, part = t >> 6;
    float s = 0.f, q = 0.f, mn = 3.402823466e38f, mx = -3.402823466e38f;
    size_t p0 = (size_t)part * 16384;
    for (size_t p = 0; p < 16384; ++p) {
      float v = Eb[(p0 + p) * 64 + c];
      s += v; q += v * v; mn = fminf(mn, v); mx = fmaxf(mx, v);
    }
    eS[part * 64 + c] = s; eQ[part * 64 + c] = q;
    eMn[part * 64 + c] = mn; eMx[part * 64 + c] = mx;
  }
  __syncthreads();
  if (t < 64) {
    float s = 0.f, q = 0.f, mn = 3.402823466e38f, mx = -3.402823466e38f;
    for (int p2 = 0; p2 < 4; ++p2) {
      s += eS[p2 * 64 + t]; q += eQ[p2 * 64 + t];
      mn = fminf(mn, eMn[p2 * 64 + t]); mx = fmaxf(mx, eMx[p2 * 64 + t]);
    }
    float mu_ = s / 65536.f;
    float var = (q - 65536.f * mu_ * mu_) / 65535.f;
    fe[t] = mu_; fe[64 + t] = mn; fe[128 + t] = mx;
    fe[192 + t] = sqrtf(fmaxf(var, 0.f));
  }
  __syncthreads();
  if (t < 64) {  // new_y = y@Wyy + feats_x@Wxy + feats_e@Wey (+biases)
    float s = byy[t] + bxy[t] + bey[t];
    for (int k2 = 0; k2 < 64; ++k2)   s += yIn[b * 64 + k2] * Wyy[k2 * 64 + t];
    for (int k2 = 0; k2 < 1024; ++k2) s += fx[k2] * Wxy[k2 * 64 + t];
    for (int k2 = 0; k2 < 256; ++k2)  s += fe[k2] * Wey[k2 * 64 + t];
    ny[t] = s;
  }
  __syncthreads();
  if (t < 64) {
    float s = bo1[t];
    for (int k2 = 0; k2 < 64; ++k2) s += ny[k2] * Wo1[k2 * 64 + t];
    tmp64[t] = fmaxf(s, 0.f);
  }
  __syncthreads();
  if (t < 64) {
    float s = bo2[t];
    for (int k2 = 0; k2 < 64; ++k2) s += tmp64[k2] * Wo2[k2 * 64 + t];
    ny[t] = yIn[b * 64 + t] + s;  // residual input to norm_y1
  }
  __syncthreads();
  if (t == 0) {
    float S = 0.f;
    for (int k2 = 0; k2 < 64; ++k2) S += ny[k2];
    float m_ = S / 64.f, Q2 = 0.f;
    for (int k2 = 0; k2 < 64; ++k2) { float d = ny[k2] - m_; Q2 += d * d; }
    mv2[0] = m_; mv2[1] = rsqrtf(Q2 / 64.f + 1e-5f);
  }
  __syncthreads();
  if (t < 64) yrS[t] = (ny[t] - mv2[0]) * mv2[1] * g1[t] + be1[t];
  __syncthreads();
  for (int rep = 0; rep < 8; ++rep) {  // y FFN hidden (2048)
    int j = t + rep * 256;
    float s = bl1[j];
    for (int k2 = 0; k2 < 64; ++k2) s += yrS[k2] * Wl1[k2 * 2048 + j];
    hS[j] = fmaxf(s, 0.f);
  }
  __syncthreads();
  if (t < 64) {
    float s = bl2[t];
    for (int k2 = 0; k2 < 2048; ++k2) s += hS[k2] * Wl2[k2 * 64 + t];
    tmp64[t] = yrS[t] + s;
  }
  __syncthreads();
  if (t == 0) {
    float S = 0.f;
    for (int k2 = 0; k2 < 64; ++k2) S += tmp64[k2];
    float m_ = S / 64.f, Q2 = 0.f;
    for (int k2 = 0; k2 < 64; ++k2) { float d = tmp64[k2] - m_; Q2 += d * d; }
    mv2[0] = m_; mv2[1] = rsqrtf(Q2 / 64.f + 1e-5f);
  }
  __syncthreads();
  if (t < 64) yo[b * 64 + t] = (tmp64[t] - mv2[0]) * mv2[1] * g2[t] + be2[t];

  // modulation vectors
  if (t < 8) {
    float s = byea[t];
    for (int k2 = 0; k2 < 64; ++k2) s += yIn[b * 64 + k2] * Wyea[k2 * 8 + t];
    yvec[YV_YEA + b * 8 + t] = s;
  }
  if (t >= 8 && t < 16) {
    int h2 = t - 8;
    float s = byem[h2];
    for (int k2 = 0; k2 < 64; ++k2) s += yIn[b * 64 + k2] * Wyem[k2 * 8 + h2];
    yvec[YV_YEM + b * 8 + h2] = s;
  }
  {
    float s = byxa[t];
    for (int k2 = 0; k2 < 64; ++k2) s += yIn[b * 64 + k2] * Wyxa[k2 * 256 + t];
    yvec[YV_YXA + b * 256 + t] = s;
  }
  {
    float s = byxm[t];
    for (int k2 = 0; k2 < 64; ++k2) s += yIn[b * 64 + k2] * Wyxm[k2 * 256 + t];
    yvec[YV_YXM + b * 256 + t] = s;
  }
  if (t == 0) {
    int idx = (int)(t_init[b] * 1000.0f);
    yvec[YV_TC + b] = alphas[idx];
  }
}

// ---------------------------------------------------------------------------
// QKV projection: out = (X @ W + bias) * node_mask(row).  M=2048,K=256,N=256.
__global__ __launch_bounds__(256) void k_qkv(const float* X, const float* W,
                                             const float* bias,
                                             const unsigned char* nm,
                                             float* out) {
  int wave = threadIdx.x >> 5, lane = threadIdx.x & 31;
  int l16 = lane & 15, hf = lane >> 4;
  int gw = blockIdx.x * 8 + wave;
  int tn = gw & 15, tm = gw >> 4;
  int m0 = tm * 16, n0 = tn * 16;
  v8f acc = {};
  for (int kk = 0; kk < 256; kk += 32) {
    v16bf a = ldA(X, m0 + l16, 256, kk, hf);
    v16bf b = ldB(W, n0 + l16, 256, kk, hf);
    acc = wmma_bf16(a, b, acc);
  }
  int col = n0 + l16;
  float bc = bias[col];
#pragma unroll
  for (int r = 0; r < 8; ++r) {
    int row = m0 + r + 8 * hf;
    float mv = nm[row] ? 1.f : 0.f;
    out[(size_t)row * 256 + col] = (acc[r] + bc) * mv;
  }
}

// ---------------------------------------------------------------------------
// E12: out[pos, 0:8]=e_mul heads, [8:16]=e_add heads, masked.  K=64, N=16.
// The block's 128x64 f32 tile of src is staged to LDS via the Tensor Data
// Mover (one 2-D descriptor per block), then WMMA A-fragments are gathered
// from LDS.  isreg!=0 additionally zeroes rows where src[pos*64]==0.
__global__ __launch_bounds__(256) void k_e12(const float* src, const float* Wm,
                                             const float* bm, const float* Wa,
                                             const float* ba,
                                             const unsigned char* nm,
                                             float* out12, int isreg) {
  __shared__ float tileA[128 * 64];  // 32 KB staged tile
  int wave = threadIdx.x >> 5, lane = threadIdx.x & 31;
  int l16 = lane & 15, hf = lane >> 4;
  int r0 = blockIdx.x * 128;  // 4096 blocks x 128 rows
#if HAVE_TDM
  if (threadIdx.x < 32) {
    // D# group 0: count=1 | lds_addr[63:32] | global_addr[120:64] | type=2
    unsigned ldsAddr = (unsigned)(size_t)(void*)tileA;  // flat->LDS: low 32 bits
    unsigned long long ga =
        (unsigned long long)(const char*)src + (unsigned long long)r0 * 256ull;
    uint4v g0;
    g0[0] = 1u;                                   // count = 1 valid descriptor
    g0[1] = ldsAddr;                              // LDS byte address
    g0[2] = (unsigned)(ga & 0xFFFFFFFFull);       // global_addr[31:0]
    g0[3] = (unsigned)((ga >> 32) & 0x01FFFFFFull) | (2u << 30);  // [56:32]|type
    // D# group 1: data_size=4B, tensor 64 x 128, tile 64 x 128, stride0=64
    int8v g1;
    g1[0] = 0x00020000;          // data_size = 2 (4 bytes)
    g1[1] = (int)(64u << 16);    // tensor_dim0 = 64 (bits 79:48)
    g1[2] = (int)(128u << 16);   // tensor_dim1 = 128 (bits 111:80)
    g1[3] = (int)(64u << 16);    // tile_dim0 = 64 (bits 127:112)
    g1[4] = 128;                 // tile_dim1 = 128, tile_dim2 = 0
    g1[5] = 64;                  // tensor_dim0_stride = 64 (elements)
    g1[6] = 0;
    g1[7] = 0;
    int4v zz = {};
#if __clang_major__ >= 23
    int8v zz8 = {};
    __builtin_amdgcn_tensor_load_to_lds(g0, g1, zz, zz, zz8, 0);
#else
    __builtin_amdgcn_tensor_load_to_lds(g0, g1, zz, zz, 0);
#endif
    __builtin_amdgcn_s_wait_tensorcnt(0);
  }
  __syncthreads();
  const float* Asrc = tileA;                      // fragments gathered from LDS
#else
  const float* Asrc = src + (size_t)r0 * 64;      // fallback: direct global
#endif
  int arow = wave * 16;
  v8f acc = {};
  for (int kk = 0; kk < 64; kk += 32) {
    v16bf a = ldA(Asrc, arow + l16, 64, kk, hf);
    v16bf bb;
#pragma unroll
    for (int e = 0; e < 16; ++e) {
      int k2 = kk + (hf << 4) + e;
      float wv = (l16 < 8) ? Wm[k2 * 8 + l16] : Wa[k2 * 8 + (l16 - 8)];
      bb[e] = (__bf16)wv;
    }
    acc = wmma_bf16(a, bb, acc);
  }
  float bc = (l16 < 8) ? bm[l16] : ba[l16 - 8];
#pragma unroll
  for (int r = 0; r < 8; ++r) {
    int lr = arow + r + 8 * hf;
    int row = r0 + lr;
    int b_ = row >> 16, rem = row & 65535, i = rem >> 8, j = rem & 255;
    float mv = (nm[b_ * 256 + i] && nm[b_ * 256 + j]) ? 1.f : 0.f;
    float v = (acc[r] + bc) * mv;
    if (isreg && Asrc[(size_t)lr * 64] == 0.f) v = 0.f;
    out12[(size_t)row * 16 + l16] = v;
  }
}

// ---------------------------------------------------------------------------
// Attention logits + E modulation + newE head construction.
__global__ __launch_bounds__(256) void k_logits(
    const float* Q, const float* K, const float* E12, const float* Er12,
    const float* Ereg, const float* yvec, float* Yl, float* NEH) {
  int wave = threadIdx.x >> 5, lane = threadIdx.x & 31;
  int l16 = lane & 15, hf = lane >> 4;
  int gw = blockIdx.x * 8 + wave;  // 16384 waves
  int jt = gw & 15, it = (gw >> 4) & 15, h = (gw >> 8) & 7, b = gw >> 11;
  int i0 = it * 16, j0 = jt * 16;
  const float* Qb = Q + (size_t)b * 65536;
  const float* Kb = K + (size_t)b * 65536;
  v16bf a = ldA(Qb, i0 + l16, 256, h * 32, hf);
  v16bf bb = ldBt(Kb, j0 + l16, 256, h * 32, hf);
  v8f acc = {};
  acc = wmma_bf16(a, bb, acc);
  const float inv = 0.17677669529663687f;  // 1/sqrt(32)
  float yea = yvec[YV_YEA + b * 8 + h];
  float yem = yvec[YV_YEM + b * 8 + h];
  float tc = yvec[YV_TC + b];
  int j = j0 + l16;
#pragma unroll
  for (int r = 0; r < 8; ++r) {
    int i = i0 + r + 8 * hf;
    size_t pos = ((size_t)(b * 256 + i)) * 256 + j;
    float e1 = E12[pos * 16 + h], e2 = E12[pos * 16 + 8 + h];
    float yv = acc[r] * inv;
    yv = yv * (e1 + 1.f) + e2;
    Yl[((size_t)(b * 8 + h) * 256 + i) * 256 + j] = yv;
    float ne = yea + (yem + 1.f) * yv;
    if (Ereg[pos * 64] != 0.f) {
      float r1 = Er12[pos * 16 + h], r2 = Er12[pos * 16 + 8 + h];
      ne = tc * ne * r1 + (1.f - tc) * ne + tc * r2;
    }
    NEH[pos * 8 + h] = ne;
  }
}

// ---------------------------------------------------------------------------
// Softmax over j (masked by node_mask[b,j]), in place on Yl (b,h,i,j).
__global__ __launch_bounds__(256) void k_softmax(float* Yl,
                                                 const unsigned char* nm) {
  int t = threadIdx.x;
  size_t rowbase = (size_t)blockIdx.x * 256;
  int b = blockIdx.x >> 11;
  __shared__ float red[256];
  float v = nm[b * 256 + t] ? Yl[rowbase + t] : -3.402823466e38f;
  red[t] = v;
  __syncthreads();
  for (int s = 128; s > 0; s >>= 1) {
    if (t < s) red[t] = fmaxf(red[t], red[t + s]);
    __syncthreads();
  }
  float mx = red[0];
  __syncthreads();
  float e = expf(v - mx);
  red[t] = e;
  __syncthreads();
  for (int s = 128; s > 0; s >>= 1) {
    if (t < s) red[t] += red[t + s];
    __syncthreads();
  }
  Yl[rowbase + t] = e / red[0];
}

// ---------------------------------------------------------------------------
// wV = attn @ V, modulated by y_x vectors -> NEWX.
__global__ __launch_bounds__(256) void k_wv(const float* attn, const float* V,
                                            const float* yvec, float* NEWX) {
  int wave = threadIdx.x >> 5, lane = threadIdx.x & 31;
  int l16 = lane & 15, hf = lane >> 4;
  int gw = blockIdx.x * 8 + wave;  // 2048 waves
  int dt = gw & 1, it = (gw >> 1) & 15, h = (gw >> 5) & 7, b = gw >> 8;
  int i0 = it * 16, d0 = dt * 16;
  const float* aP = attn + (size_t)(b * 8 + h) * 65536;
  const float* vP = V + (size_t)b * 65536 + h * 32 + d0;
  v8f acc = {};
  for (int kk = 0; kk < 256; kk += 32) {
    v16bf a = ldA(aP, i0 + l16, 256, kk, hf);
    v16bf bb = ldB(vP, l16, 256, kk, hf);
    acc = wmma_bf16(a, bb, acc);
  }
  int c = h * 32 + d0 + l16;
  float xa = yvec[YV_YXA + b * 256 + c];
  float xm_ = yvec[YV_YXM + b * 256 + c];
#pragma unroll
  for (int r = 0; r < 8; ++r) {
    int i = i0 + r + 8 * hf;
    NEWX[(size_t)(b * 256 + i) * 256 + c] = xa + (xm_ + 1.f) * acc[r];
  }
}

// ---------------------------------------------------------------------------
// Xr = ln(normX1, X + (NEWX @ Wxout + b)*xm).  Block = 16 rows x 256 cols.
__global__ __launch_bounds__(256) void k_xout(const float* NEWX, const float* W,
                                              const float* bias, const float* X,
                                              const unsigned char* nm,
                                              const float* gam, const float* bet,
                                              float* Xr) {
  __shared__ float tileS[16 * 256];
  __shared__ float red[256], red2[256];
  __shared__ float mu[16], rs[16];
  int wave = threadIdx.x >> 5, lane = threadIdx.x & 31;
  int l16 = lane & 15, hf = lane >> 4;
  int r0 = blockIdx.x * 16;
  for (int tt = 0; tt < 2; ++tt) {
    int n0 = (wave * 2 + tt) * 16;
    v8f acc = {};
    for (int kk = 0; kk < 256; kk += 32) {
      v16bf a = ldA(NEWX, r0 + l16, 256, kk, hf);
      v16bf b = ldB(W, n0 + l16, 256, kk, hf);
      acc = wmma_bf16(a, b, acc);
    }
    int col = n0 + l16;
    float bc = bias[col];
#pragma unroll
    for (int r = 0; r < 8; ++r) {
      int m = r + 8 * hf, row = r0 + m;
      float mv = nm[row] ? 1.f : 0.f;
      tileS[m * 256 + col] = X[(size_t)row * 256 + col] + (acc[r] + bc) * mv;
    }
  }
  __syncthreads();
  int t = threadIdx.x, rrow = t >> 4, lc = t & 15;
  float s = 0.f, q = 0.f;
  for (int rep = 0; rep < 16; ++rep) {
    float v = tileS[rrow * 256 + lc + rep * 16];
    s += v; q += v * v;
  }
  red[t] = s; red2[t] = q;
  __syncthreads();
  if (t < 16) {
    float S = 0.f, Q2 = 0.f;
    for (int k2 = 0; k2 < 16; ++k2) { S += red[t * 16 + k2]; Q2 += red2[t * 16 + k2]; }
    float m_ = S / 256.f;
    mu[t] = m_; rs[t] = rsqrtf(Q2 / 256.f - m_ * m_ + 1e-5f);
  }
  __syncthreads();
  for (int rep = 0; rep < 16; ++rep) {
    int col = lc + rep * 16;
    float v = tileS[rrow * 256 + col];
    Xr[(size_t)(r0 + rrow) * 256 + col] =
        (v - mu[rrow]) * rs[rrow] * gam[col] + bet[col];
  }
}

// ---------------------------------------------------------------------------
// Er = ln(normE1, E + (NEH @ We_out + b)*em) -> written into Eo region.
__global__ __launch_bounds__(128) void k_eout(const float* NEH, const float* W,
                                              const float* bias, const float* E,
                                              const unsigned char* nm,
                                              const float* gam, const float* bet,
                                              float* Er) {
  __shared__ float tileS[16 * 64];
  __shared__ float red[128], red2[128];
  __shared__ float mu[16], rs[16];
  int wave = threadIdx.x >> 5, lane = threadIdx.x & 31;
  int l16 = lane & 15, hf = lane >> 4;
  int r0 = blockIdx.x * 16;
  int n0 = wave * 16;
  v16bf a, bb;
#pragma unroll
  for (int g2 = 0; g2 < 2; ++g2)
#pragma unroll
    for (int e = 0; e < 8; ++e) {
      int k2 = 16 * hf + 8 * g2 + e;  // A mapping
      a[g2 * 8 + e] =
          (k2 < 8) ? (__bf16)NEH[(size_t)(r0 + l16) * 8 + k2] : (__bf16)0.f;
    }
#pragma unroll
  for (int e = 0; e < 16; ++e) {
    int k2 = 16 * hf + e;  // B mapping
    bb[e] = (k2 < 8) ? (__bf16)W[k2 * 64 + n0 + l16] : (__bf16)0.f;
  }
  v8f acc = {};
  acc = wmma_bf16(a, bb, acc);
  int col = n0 + l16;
  float bc = bias[col];
#pragma unroll
  for (int r = 0; r < 8; ++r) {
    int m = r + 8 * hf, row = r0 + m;
    int b_ = row >> 16, rem = row & 65535, i = rem >> 8, j = rem & 255;
    float mv = (nm[b_ * 256 + i] && nm[b_ * 256 + j]) ? 1.f : 0.f;
    tileS[m * 64 + col] = E[(size_t)row * 64 + col] + (acc[r] + bc) * mv;
  }
  __syncthreads();
  int t = threadIdx.x, rrow = t >> 3, lc = t & 7;
  float s = 0.f, q = 0.f;
  for (int rep = 0; rep < 8; ++rep) {
    float v = tileS[rrow * 64 + lc + rep * 8];
    s += v; q += v * v;
  }
  red[t] = s; red2[t] = q;
  __syncthreads();
  if (t < 16) {
    float S = 0.f, Q2 = 0.f;
    for (int k2 = 0; k2 < 8; ++k2) { S += red[t * 8 + k2]; Q2 += red2[t * 8 + k2]; }
    float m_ = S / 64.f;
    mu[t] = m_; rs[t] = rsqrtf(Q2 / 64.f - m_ * m_ + 1e-5f);
  }
  __syncthreads();
  for (int rep = 0; rep < 8; ++rep) {
    int col2 = lc + rep * 8;
    float v = tileS[rrow * 64 + col2];
    Er[(size_t)(r0 + rrow) * 64 + col2] =
        (v - mu[rrow]) * rs[rrow] * gam[col2] + bet[col2];
  }
}

// ---------------------------------------------------------------------------
// E FFN fused, in place: Eio = ln(normE2, Eio + lin2(relu(lin1(Eio)))).
__global__ __launch_bounds__(256) void k_effn(float* Eio, const float* W1,
                                              const float* b1, const float* W2,
                                              const float* b2, const float* gam,
                                              const float* bet) {
  __shared__ float hidS[16 * 128];
  __shared__ float outS[16 * 64];
  __shared__ float red[256], red2[256];
  __shared__ float mu[16], rs[16];
  int wave = threadIdx.x >> 5, lane = threadIdx.x & 31;
  int l16 = lane & 15, hf = lane >> 4;
  int r0 = blockIdx.x * 16;
  {  // stage 1: hidden = relu(Er @ W1 + b1)  (16x128)
    int n0 = wave * 16;
    v8f acc = {};
    for (int kk = 0; kk < 64; kk += 32) {
      v16bf a = ldA(Eio, r0 + l16, 64, kk, hf);
      v16bf b = ldB(W1, n0 + l16, 128, kk, hf);
      acc = wmma_bf16(a, b, acc);
    }
    int col = n0 + l16;
    float bc = b1[col];
#pragma unroll
    for (int r = 0; r < 8; ++r) {
      int m = r + 8 * hf;
      hidS[m * 128 + col] = fmaxf(acc[r] + bc, 0.f);
    }
  }
  __syncthreads();
  if (wave < 4) {  // stage 2: out = Er + hidden @ W2 + b2  (16x64)
    int n0 = wave * 16;
    v8f acc = {};
    for (int kk = 0; kk < 128; kk += 32) {
      v16bf a = ldA(hidS, l16, 128, kk, hf);
      v16bf b = ldB(W2, n0 + l16, 64, kk, hf);
      acc = wmma_bf16(a, b, acc);
    }
    int col = n0 + l16;
    float bc = b2[col];
#pragma unroll
    for (int r = 0; r < 8; ++r) {
      int m = r + 8 * hf, row = r0 + m;
      outS[m * 64 + col] = Eio[(size_t)row * 64 + col] + acc[r] + bc;
    }
  }
  __syncthreads();
  int t = threadIdx.x, rrow = t >> 4, lc = t & 15;
  float s = 0.f, q = 0.f;
  for (int rep = 0; rep < 4; ++rep) {
    float v = outS[rrow * 64 + lc + rep * 16];
    s += v; q += v * v;
  }
  red[t] = s; red2[t] = q;
  __syncthreads();
  if (t < 16) {
    float S = 0.f, Q2 = 0.f;
    for (int k2 = 0; k2 < 16; ++k2) { S += red[t * 16 + k2]; Q2 += red2[t * 16 + k2]; }
    float m_ = S / 64.f;
    mu[t] = m_; rs[t] = rsqrtf(Q2 / 64.f - m_ * m_ + 1e-5f);
  }
  __syncthreads();
  for (int rep = 0; rep < 4; ++rep) {
    int col = lc + rep * 16;
    float v = outS[rrow * 64 + col];
    Eio[(size_t)(r0 + rrow) * 64 + col] =
        (v - mu[rrow]) * rs[rrow] * gam[col] + bet[col];
  }
}

// ---------------------------------------------------------------------------
// X FFN fused: Xo = ln(normX2, Xr + lin2(relu(lin1(Xr)))).
__global__ __launch_bounds__(256) void k_xffn(const float* Xr, const float* W1,
                                              const float* b1, const float* W2,
                                              const float* b2, const float* gam,
                                              const float* bet, float* Xo) {
  __shared__ __bf16 hidH[16 * 256];
  __shared__ float outS[16 * 256];
  __shared__ float red[256], red2[256];
  __shared__ float mu[16], rs[16];
  int wave = threadIdx.x >> 5, lane = threadIdx.x & 31;
  int l16 = lane & 15, hf = lane >> 4;
  int r0 = blockIdx.x * 16;
  v8f accO[2] = {{}, {}};
  for (int ch = 0; ch < 8; ++ch) {
#pragma unroll
    for (int tt = 0; tt < 2; ++tt) {  // stage 1: hidden chunk
      int n0 = (wave * 2 + tt) * 16;
      v8f h = {};
      for (int kk = 0; kk < 256; kk += 32) {
        v16bf a = ldA(Xr, r0 + l16, 256, kk, hf);
        v16bf b = ldB(W1 + ch * 256, n0 + l16, 2048, kk, hf);
        h = wmma_bf16(a, b, h);
      }
      int col = n0 + l16;
      float bc = b1[ch * 256 + col];
#pragma unroll
      for (int r = 0; r < 8; ++r) {
        int m = r + 8 * hf;
        hidH[m * 256 + col] = (__bf16)fmaxf(h[r] + bc, 0.f);
      }
    }
    __syncthreads();
#pragma unroll
    for (int tt = 0; tt < 2; ++tt) {  // stage 2: accumulate output tiles
      int n0 = (wave * 2 + tt) * 16;
      for (int kk = 0; kk < 256; kk += 32) {
        v16bf a = ldA_h(hidH, l16, 256, kk, hf);
        v16bf b = ldB(W2 + (size_t)ch * 256 * 256, n0 + l16, 256, kk, hf);
        accO[tt] = wmma_bf16(a, b, accO[tt]);
      }
    }
    __syncthreads();
  }
#pragma unroll
  for (int tt = 0; tt < 2; ++tt) {
    int n0 = (wave * 2 + tt) * 16, col = n0 + l16;
    float bc = b2[col];
#pragma unroll
    for (int r = 0; r < 8; ++r) {
      int m = r + 8 * hf;
      outS[m * 256 + col] =
          Xr[(size_t)(r0 + m) * 256 + col] + accO[tt][r] + bc;
    }
  }
  __syncthreads();
  int t = threadIdx.x, rrow = t >> 4, lc = t & 15;
  float s = 0.f, q = 0.f;
  for (int rep = 0; rep < 16; ++rep) {
    float v = outS[rrow * 256 + lc + rep * 16];
    s += v; q += v * v;
  }
  red[t] = s; red2[t] = q;
  __syncthreads();
  if (t < 16) {
    float S = 0.f, Q2 = 0.f;
    for (int k2 = 0; k2 < 16; ++k2) { S += red[t * 16 + k2]; Q2 += red2[t * 16 + k2]; }
    float m_ = S / 256.f;
    mu[t] = m_; rs[t] = rsqrtf(Q2 / 256.f - m_ * m_ + 1e-5f);
  }
  __syncthreads();
  for (int rep = 0; rep < 16; ++rep) {
    int col = lc + rep * 16;
    float v = outS[rrow * 256 + col];
    Xo[(size_t)(r0 + rrow) * 256 + col] =
        (v - mu[rrow]) * rs[rrow] * gam[col] + bet[col];
  }
}

// ---------------------------------------------------------------------------
extern "C" void kernel_launch(void* const* d_in, const int* in_sizes, int n_in,
                              void* d_out, int out_size, void* d_ws,
                              size_t ws_size, hipStream_t stream) {
  (void)in_sizes; (void)out_size;
  if (n_in < 67) return;
  // Params are the sorted-pytree leaves of the params dict, starting at index 7,
  // each entry contributing (b, w) or (b, g) in that order.
  enum {
    I_E_ADD = 0, I_E_MUL, I_E_OUT, I_E_REG_ADD, I_E_REG_MUL, I_E_Y, I_K,
    I_LINE1, I_LINE2, I_LINX1, I_LINX2, I_LIN_Y1, I_LIN_Y2,
    I_NORME1, I_NORME2, I_NORMX1, I_NORMX2, I_NORM_Y1, I_NORM_Y2,
    I_Q, I_V, I_X_OUT, I_X_Y, I_Y_E_ADD, I_Y_E_MUL, I_Y_OUT1, I_Y_OUT2,
    I_Y_X_ADD, I_Y_X_MUL, I_Y_Y
  };
  auto Pb = [&](int i) { return (const float*)d_in[7 + 2 * i]; };
  auto Pw = [&](int i) { return (const float*)d_in[7 + 2 * i + 1]; };

  const float* X = (const float*)d_in[0];
  const float* E = (const float*)d_in[1];
  const float* y = (const float*)d_in[2];
  const float* t_init = (const float*)d_in[3];
  const unsigned char* nm = (const unsigned char*)d_in[4];
  const float* Ereg = (const float*)d_in[5];
  const float* alphas = (const float*)d_in[6];

  // Workspace layout (floats)
  const size_t needed = (size_t)27795456 * 4;
  if (ws_size < needed) return;
  float* ws = (float*)d_ws;
  float* Qb   = ws + 0;         // 524288
  float* Kb   = ws + 524288;    // 524288
  float* Vb   = ws + 1048576;   // 524288
  float* E12  = ws + 1572864;   // 8388608
  float* Er12 = ws + 9961472;   // 8388608
  float* Yl   = ws + 18350080;  // 4194304
  float* NEH  = ws + 22544384;  // 4194304
  float* NEWX = ws + 26738688;  // 524288
  float* XrB  = ws + 27262976;  // 524288
  float* yvec = ws + 27787264;  // 8192

  float* out = (float*)d_out;
  float* Xo = out;
  float* Eo = out + 524288;        // also holds Er between k_eout and k_effn
  float* yo = out + 34078720;

  k_ypath<<<8, 256, 0, stream>>>(
      X, E, y, t_init, alphas, Pw(I_Y_Y), Pb(I_Y_Y), Pw(I_X_Y), Pb(I_X_Y),
      Pw(I_E_Y), Pb(I_E_Y), Pw(I_Y_OUT1), Pb(I_Y_OUT1), Pw(I_Y_OUT2),
      Pb(I_Y_OUT2), Pw(I_NORM_Y1), Pb(I_NORM_Y1), Pw(I_LIN_Y1), Pb(I_LIN_Y1),
      Pw(I_LIN_Y2), Pb(I_LIN_Y2), Pw(I_NORM_Y2), Pb(I_NORM_Y2), Pw(I_Y_E_ADD),
      Pb(I_Y_E_ADD), Pw(I_Y_E_MUL), Pb(I_Y_E_MUL), Pw(I_Y_X_ADD), Pb(I_Y_X_ADD),
      Pw(I_Y_X_MUL), Pb(I_Y_X_MUL), yo, yvec);

  k_qkv<<<256, 256, 0, stream>>>(X, Pw(I_Q), Pb(I_Q), nm, Qb);
  k_qkv<<<256, 256, 0, stream>>>(X, Pw(I_K), Pb(I_K), nm, Kb);
  k_qkv<<<256, 256, 0, stream>>>(X, Pw(I_V), Pb(I_V), nm, Vb);

  k_e12<<<4096, 256, 0, stream>>>(E, Pw(I_E_MUL), Pb(I_E_MUL), Pw(I_E_ADD),
                                  Pb(I_E_ADD), nm, E12, 0);
  k_e12<<<4096, 256, 0, stream>>>(Ereg, Pw(I_E_REG_MUL), Pb(I_E_REG_MUL),
                                  Pw(I_E_REG_ADD), Pb(I_E_REG_ADD), nm, Er12, 1);

  k_logits<<<2048, 256, 0, stream>>>(Qb, Kb, E12, Er12, Ereg, yvec, Yl, NEH);
  k_softmax<<<16384, 256, 0, stream>>>(Yl, nm);
  k_wv<<<256, 256, 0, stream>>>(Yl, Vb, yvec, NEWX);

  k_xout<<<128, 256, 0, stream>>>(NEWX, Pw(I_X_OUT), Pb(I_X_OUT), X, nm,
                                  Pw(I_NORMX1), Pb(I_NORMX1), XrB);
  k_eout<<<32768, 128, 0, stream>>>(NEH, Pw(I_E_OUT), Pb(I_E_OUT), E, nm,
                                    Pw(I_NORME1), Pb(I_NORME1), Eo);

  k_xffn<<<128, 256, 0, stream>>>(XrB, Pw(I_LINX1), Pb(I_LINX1), Pw(I_LINX2),
                                  Pb(I_LINX2), Pw(I_NORMX2), Pb(I_NORMX2), Xo);
  k_effn<<<32768, 256, 0, stream>>>(Eo, Pw(I_LINE1), Pb(I_LINE1), Pw(I_LINE2),
                                    Pb(I_LINE2), Pw(I_NORME2), Pb(I_NORME2));
}
